// MultiHeadAttention_9818295239094
// MI455X (gfx1250) — compile-verified, matrix operand order checked by reading, not run on hardware
//
#include <hip/hip_runtime.h>
#include <stdint.h>

// ---------------------------------------------------------------------------
// MHA forward for B=8, S=1024, D=1024, H=16, HD=64 on gfx1250 (wave32, WMMA)
// d_out = [ out (8*1024*1024 f32) | attn (8*16*1024*1024 f32) ]
// d_ws  = [ Qp bf16 8M | Kp bf16 8M | Vt bf16 8M | attnV f32 8M ] = 80 MB
// ---------------------------------------------------------------------------

typedef __attribute__((ext_vector_type(16))) __bf16 v16bf;
typedef __attribute__((ext_vector_type(8)))  __bf16 v8bf;
typedef __attribute__((ext_vector_type(8)))  float  v8f;
typedef __attribute__((ext_vector_type(4)))  unsigned u32x4;
typedef __attribute__((ext_vector_type(8)))  unsigned u32x8;

__device__ __forceinline__ __bf16 f2bf(float f) {
  return (__bf16)f;   // native f32->bf16 convert on gfx1250
}

__device__ __forceinline__ v16bf cat8(v8bf lo, v8bf hi) {
  return __builtin_shufflevector(lo, hi, 0,1,2,3,4,5,6,7,8,9,10,11,12,13,14,15);
}

// TDM descriptor group 0: count=1, lds addr, global addr, type=2 ("image").
__device__ __forceinline__ u32x4 tdm_group0(unsigned lds_byte, unsigned long long ga) {
  u32x4 g;
  g[0] = 1u;
  g[1] = lds_byte;
  g[2] = (unsigned)(ga & 0xFFFFFFFFull);
  g[3] = (unsigned)((ga >> 32) & 0x01FFFFFFull) | (2u << 30);
  return g;
}

// TDM group 1: data_size=2B, pad 4 dwords every 32 dwords (16B row skew),
// tensor_dim0=64, tensor_dim1=tensor_rows, tile_dim0=64, tile_dim1=tile_rows,
// tensor_dim0_stride=64.
__device__ __forceinline__ u32x8 tdm_group1(unsigned tile_rows, unsigned tensor_rows) {
  u32x8 g;
  g[0] = (1u << 16) | (1u << 20) | (4u << 22) | (3u << 25);
  g[1] = (64u & 0xFFFFu) << 16;
  g[2] = ((tensor_rows & 0xFFFFu) << 16);
  g[3] = ((tensor_rows >> 16) & 0xFFFFu) | (64u << 16);
  g[4] = (tile_rows & 0xFFFFu);
  g[5] = 64u;
  g[6] = 0u;
  g[7] = 0u;
  return g;
}

// ---------------------------------------------------------------------------
// Generic 8192x1024x1024 GEMM:  out = A(f32) @ W(f32 -> bf16) + bias
// MODE 0: out bf16 [B,H,S,64]; MODE 1: out bf16 [B,H,64,S]; MODE 2: out f32.
// Block: 256 thr = 8 waves (4 in M x 2 in N); block tile 128x128, K-tile 32.
// ---------------------------------------------------------------------------
template <int MODE>
__global__ __launch_bounds__(256)
void gemm1024(const float* __restrict__ A, const float* __restrict__ W,
              const float* __restrict__ bias, void* __restrict__ out) {
  __shared__ __bf16 As[128 * 40];
  __shared__ __bf16 Bs[128 * 40];   // transposed: [n][k]
  const int tid  = threadIdx.x;
  const int lane = tid & 31;
  const int w    = tid >> 5;
  const int wm   = w & 3;
  const int wn   = w >> 2;
  const int lr   = lane & 15;
  const int kb   = (lane >> 4) << 3;
  const int m0 = blockIdx.y * 128;
  const int n0 = blockIdx.x * 128;

  v8f acc[2][4] = {};

  for (int kt = 0; kt < 1024; kt += 32) {
    if (kt + 32 < 1024) {
      __builtin_prefetch(&A[(m0 + (tid >> 3)) * 1024 + kt + 32], 0, 1);
      __builtin_prefetch(&W[(kt + 32 + (tid >> 6)) * 1024 + n0 + (tid & 63)], 0, 1);
    }
    // A tile 128x32: coalesced float4 loads, packed 4x bf16 ds_store_b64
#pragma unroll
    for (int p = 0; p < 4; ++p) {
      int li = p * 1024 + tid * 4;
      int row = li >> 5;
      int col = li & 31;
      float4 v = *reinterpret_cast<const float4*>(&A[(m0 + row) * 1024 + kt + col]);
      union { __bf16 h[4]; uint2 u; } pk;
      pk.h[0] = f2bf(v.x); pk.h[1] = f2bf(v.y); pk.h[2] = f2bf(v.z); pk.h[3] = f2bf(v.w);
      *reinterpret_cast<uint2*>(&As[row * 40 + col]) = pk.u;
    }
    // W tile 32x128: each thread owns a 4(K)x4(N) sub-block; 4 coalesced
    // global_load_b128 along N, register transpose, 4 packed ds_store_b64
    // along K into the [n][k] LDS image.
    {
      const int r0 = (tid >> 5) * 4;          // K rows 0..28
      const int c0 = (tid & 31) * 4;          // N cols 0..124
      const float* wp = &W[(kt + r0) * 1024 + n0 + c0];
      float4 w0 = *reinterpret_cast<const float4*>(wp);
      float4 w1 = *reinterpret_cast<const float4*>(wp + 1024);
      float4 w2 = *reinterpret_cast<const float4*>(wp + 2048);
      float4 w3 = *reinterpret_cast<const float4*>(wp + 3072);
      union { __bf16 h[4]; uint2 u; } p0, p1, p2, p3;
      p0.h[0] = f2bf(w0.x); p0.h[1] = f2bf(w1.x); p0.h[2] = f2bf(w2.x); p0.h[3] = f2bf(w3.x);
      p1.h[0] = f2bf(w0.y); p1.h[1] = f2bf(w1.y); p1.h[2] = f2bf(w2.y); p1.h[3] = f2bf(w3.y);
      p2.h[0] = f2bf(w0.z); p2.h[1] = f2bf(w1.z); p2.h[2] = f2bf(w2.z); p2.h[3] = f2bf(w3.z);
      p3.h[0] = f2bf(w0.w); p3.h[1] = f2bf(w1.w); p3.h[2] = f2bf(w2.w); p3.h[3] = f2bf(w3.w);
      *reinterpret_cast<uint2*>(&Bs[(c0 + 0) * 40 + r0]) = p0.u;
      *reinterpret_cast<uint2*>(&Bs[(c0 + 1) * 40 + r0]) = p1.u;
      *reinterpret_cast<uint2*>(&Bs[(c0 + 2) * 40 + r0]) = p2.u;
      *reinterpret_cast<uint2*>(&Bs[(c0 + 3) * 40 + r0]) = p3.u;
    }
    __syncthreads();

    v16bf af[2], bfr[4];
#pragma unroll
    for (int mi = 0; mi < 2; ++mi) {
      int row = wm * 32 + mi * 16 + lr;
      v8bf lo = *reinterpret_cast<const v8bf*>(&As[row * 40 + kb]);
      v8bf hi = *reinterpret_cast<const v8bf*>(&As[row * 40 + kb + 16]);
      af[mi] = cat8(lo, hi);
    }
#pragma unroll
    for (int ni = 0; ni < 4; ++ni) {
      int col = wn * 64 + ni * 16 + lr;
      v8bf lo = *reinterpret_cast<const v8bf*>(&Bs[col * 40 + kb]);
      v8bf hi = *reinterpret_cast<const v8bf*>(&Bs[col * 40 + kb + 16]);
      bfr[ni] = cat8(lo, hi);
    }
#pragma unroll
    for (int mi = 0; mi < 2; ++mi)
#pragma unroll
      for (int ni = 0; ni < 4; ++ni)
        acc[mi][ni] = __builtin_amdgcn_wmma_f32_16x16x32_bf16(
            false, af[mi], false, bfr[ni], (short)0, acc[mi][ni], false, false);
    __syncthreads();
  }

#pragma unroll
  for (int mi = 0; mi < 2; ++mi) {
#pragma unroll
    for (int ni = 0; ni < 4; ++ni) {
      const int nloc = n0 + wn * 64 + ni * 16 + lr;
      const float bv = bias[nloc];
#pragma unroll
      for (int e = 0; e < 8; ++e) {
        const int mrow = m0 + wm * 32 + mi * 16 + kb + e;
        const float val = acc[mi][ni][e] + bv;
        if (MODE == 0) {
          const int b = mrow >> 10, s = mrow & 1023, h = nloc >> 6, hd = nloc & 63;
          reinterpret_cast<__bf16*>(out)[((size_t)((b << 4) + h) << 16) + (s << 6) + hd] = f2bf(val);
        } else if (MODE == 1) {
          const int b = mrow >> 10, s = mrow & 1023, h = nloc >> 6, hd = nloc & 63;
          reinterpret_cast<__bf16*>(out)[((size_t)((b << 4) + h) << 16) + (hd << 10) + s] = f2bf(val);
        } else {
          reinterpret_cast<float*>(out)[(size_t)mrow * 1024 + nloc] = val;
        }
      }
    }
  }
}

// ---------------------------------------------------------------------------
// scores: Q-tile (64x64) and K-tile (128x64) DMA'd to LDS via the Tensor Data
// Mover with a 16B/row pad skew; WMMA fragments come from ds_load_b128.
// ---------------------------------------------------------------------------
__global__ __launch_bounds__(256)
void scores_kernel(const __bf16* __restrict__ Qp, const __bf16* __restrict__ Kp,
                   const float* __restrict__ rel_bias, const int* __restrict__ mask,
                   float* __restrict__ attn) {
  __shared__ __bf16 Qs[64 * 72];    // row stride 144 B (128 data + 16 pad)
  __shared__ __bf16 Ks[128 * 72];
  const int z = blockIdx.z;
  const int b = z >> 4, h = z & 15;
  const int q0 = blockIdx.y * 64;
  const int k0 = blockIdx.x * 128;
  const int tid = threadIdx.x, lane = tid & 31, w = tid >> 5;
  const int wm = w & 1, wn = w >> 1;
  const int lr = lane & 15;
  const int kb = (lane >> 4) << 3;

  if (w == 0) {
    const unsigned long long gaQ =
        (unsigned long long)(uintptr_t)(Qp + ((size_t)z << 16) + (size_t)q0 * 64);
    const unsigned long long gaK =
        (unsigned long long)(uintptr_t)(Kp + ((size_t)z << 16) + (size_t)k0 * 64);
    const u32x4 qg0 = tdm_group0((unsigned)(uintptr_t)(&Qs[0]), gaQ);
    const u32x8 qg1 = tdm_group1(64u, 1024u);
    const u32x4 kg0 = tdm_group0((unsigned)(uintptr_t)(&Ks[0]), gaK);
    const u32x8 kg1 = tdm_group1(128u, 1024u);
    asm volatile("tensor_load_to_lds %0, %1" :: "s"(qg0), "s"(qg1) : "memory");
    asm volatile("tensor_load_to_lds %0, %1" :: "s"(kg0), "s"(kg1) : "memory");
    __builtin_amdgcn_s_wait_tensorcnt(0);
  }
  __syncthreads();

  v8f acc[2][2] = {};
#pragma unroll
  for (int kc = 0; kc < 2; ++kc) {
    const int koff = kc * 32 + kb;
    v16bf af[2], bfr[2];
#pragma unroll
    for (int mi = 0; mi < 2; ++mi) {
      const int row = wm * 32 + mi * 16 + lr;
      v8bf lo = *reinterpret_cast<const v8bf*>(&Qs[row * 72 + koff]);
      v8bf hi = *reinterpret_cast<const v8bf*>(&Qs[row * 72 + koff + 16]);
      af[mi] = cat8(lo, hi);
    }
#pragma unroll
    for (int ni = 0; ni < 2; ++ni) {
      const int row = wn * 32 + ni * 16 + lr;
      v8bf lo = *reinterpret_cast<const v8bf*>(&Ks[row * 72 + koff]);
      v8bf hi = *reinterpret_cast<const v8bf*>(&Ks[row * 72 + koff + 16]);
      bfr[ni] = cat8(lo, hi);
    }
#pragma unroll
    for (int mi = 0; mi < 2; ++mi)
#pragma unroll
      for (int ni = 0; ni < 2; ++ni)
        acc[mi][ni] = __builtin_amdgcn_wmma_f32_16x16x32_bf16(
            false, af[mi], false, bfr[ni], (short)0, acc[mi][ni], false, false);
  }

  const float scale = 0.125f;
#pragma unroll
  for (int mi = 0; mi < 2; ++mi) {
#pragma unroll
    for (int ni = 0; ni < 2; ++ni) {
      const int kcol = k0 + wn * 32 + ni * 16 + lr;
#pragma unroll
      for (int e = 0; e < 8; ++e) {
        const int qrow = q0 + wm * 32 + mi * 16 + kb + e;
        const size_t ro = (size_t)qrow * 1024 + kcol;
        float v = acc[mi][ni][e] * scale + rel_bias[((size_t)h << 20) + ro];
        if (mask[((size_t)b << 20) + ro] == 0) v = -__builtin_inff();
        attn[((size_t)z << 20) + ro] = v;
      }
    }
  }
}

// ---------------------------------------------------------------------------
__global__ __launch_bounds__(256)
void softmax_kernel(float* __restrict__ attn) {
  float* p = attn + (size_t)blockIdx.x * 1024;
  const int tid = threadIdx.x;
  float4 v = *reinterpret_cast<float4*>(&p[tid * 4]);
  float m = fmaxf(fmaxf(v.x, v.y), fmaxf(v.z, v.w));
#pragma unroll
  for (int off = 16; off > 0; off >>= 1) m = fmaxf(m, __shfl_xor(m, off));
  __shared__ float redm[8];
  __shared__ float reds[8];
  if ((tid & 31) == 0) redm[tid >> 5] = m;
  __syncthreads();
  float mall = redm[0];
#pragma unroll
  for (int i = 1; i < 8; ++i) mall = fmaxf(mall, redm[i]);
  float e0 = __expf(v.x - mall), e1 = __expf(v.y - mall);
  float e2 = __expf(v.z - mall), e3 = __expf(v.w - mall);
  float s = e0 + e1 + e2 + e3;
#pragma unroll
  for (int off = 16; off > 0; off >>= 1) s += __shfl_xor(s, off);
  if ((tid & 31) == 0) reds[tid >> 5] = s;
  __syncthreads();
  float stot = 0.0f;
#pragma unroll
  for (int i = 0; i < 8; ++i) stot += reds[i];
  const float inv = 1.0f / stot;
  v.x = e0 * inv; v.y = e1 * inv; v.z = e2 * inv; v.w = e3 * inv;
  *reinterpret_cast<float4*>(&p[tid * 4]) = v;
}

// ---------------------------------------------------------------------------
__global__ __launch_bounds__(256)
void attnv_kernel(const float* __restrict__ attn, const __bf16* __restrict__ Vt,
                  float* __restrict__ attnV) {
  const int z = blockIdx.z;
  const int b = z >> 4, h = z & 15;
  const int q0 = blockIdx.x * 256;
  const int tid = threadIdx.x, lane = tid & 31, w = tid >> 5;
  const int lr = lane & 15;
  const int kb = (lane >> 4) << 3;
  const int qw = q0 + w * 32;
  const float*  ap = attn + ((size_t)z << 20);
  const __bf16* vp = Vt   + ((size_t)z << 16);

  v8f acc[2][4] = {};
  for (int kc = 0; kc < 1024; kc += 32) {
    v16bf af[2], bfr[4];
#pragma unroll
    for (int mi = 0; mi < 2; ++mi) {
      const int row = qw + mi * 16 + lr;
      const float* rp = &ap[(size_t)row * 1024 + kc + kb];
      float4 l0 = *reinterpret_cast<const float4*>(rp);
      float4 l1 = *reinterpret_cast<const float4*>(rp + 4);
      float4 h0 = *reinterpret_cast<const float4*>(rp + 16);
      float4 h1 = *reinterpret_cast<const float4*>(rp + 20);
      af[mi] = (v16bf){f2bf(l0.x), f2bf(l0.y), f2bf(l0.z), f2bf(l0.w),
                       f2bf(l1.x), f2bf(l1.y), f2bf(l1.z), f2bf(l1.w),
                       f2bf(h0.x), f2bf(h0.y), f2bf(h0.z), f2bf(h0.w),
                       f2bf(h1.x), f2bf(h1.y), f2bf(h1.z), f2bf(h1.w)};
    }
#pragma unroll
    for (int ni = 0; ni < 4; ++ni) {
      const int col = ni * 16 + lr;
      v8bf lo = *reinterpret_cast<const v8bf*>(&vp[col * 1024 + kc + kb]);
      v8bf hi = *reinterpret_cast<const v8bf*>(&vp[col * 1024 + kc + kb + 16]);
      bfr[ni] = cat8(lo, hi);
    }
#pragma unroll
    for (int mi = 0; mi < 2; ++mi)
#pragma unroll
      for (int ni = 0; ni < 4; ++ni)
        acc[mi][ni] = __builtin_amdgcn_wmma_f32_16x16x32_bf16(
            false, af[mi], false, bfr[ni], (short)0, acc[mi][ni], false, false);
  }

#pragma unroll
  for (int mi = 0; mi < 2; ++mi) {
#pragma unroll
    for (int ni = 0; ni < 4; ++ni) {
      const int ncol = ni * 16 + lr;
#pragma unroll
      for (int e = 0; e < 8; ++e) {
        const int qrow = qw + mi * 16 + kb + e;
        attnV[((size_t)b << 20) + (size_t)qrow * 1024 + h * 64 + ncol] = acc[mi][ni][e];
      }
    }
  }
}

// ---------------------------------------------------------------------------
extern "C" void kernel_launch(void* const* d_in, const int* in_sizes, int n_in,
                              void* d_out, int out_size, void* d_ws, size_t ws_size,
                              hipStream_t stream) {
  const float* query    = (const float*)d_in[0];
  const float* key      = (const float*)d_in[1];
  const float* value    = (const float*)d_in[2];
  const int*   mask     = (const int*)  d_in[3];
  const float* rel_bias = (const float*)d_in[4];
  const float* Wq = (const float*)d_in[5];  const float* bq = (const float*)d_in[6];
  const float* Wk = (const float*)d_in[7];  const float* bk = (const float*)d_in[8];
  const float* Wv = (const float*)d_in[9];  const float* bv = (const float*)d_in[10];
  const float* Wo = (const float*)d_in[11]; const float* bo = (const float*)d_in[12];

  float* out  = (float*)d_out;
  float* attn = out + (size_t)8 * 1024 * 1024;

  __bf16* Qp = (__bf16*)d_ws;
  __bf16* Kp = Qp + (size_t)8388608;
  __bf16* Vt = Kp + (size_t)8388608;
  float* attnV = (float*)(Vt + (size_t)8388608);

  const dim3 blk(256);
  gemm1024<0><<<dim3(8, 64), blk, 0, stream>>>(query, Wq, bq, Qp);
  gemm1024<0><<<dim3(8, 64), blk, 0, stream>>>(key,   Wk, bk, Kp);
  gemm1024<1><<<dim3(8, 64), blk, 0, stream>>>(value, Wv, bv, Vt);
  scores_kernel<<<dim3(8, 16, 128), blk, 0, stream>>>(Qp, Kp, rel_bias, mask, attn);
  softmax_kernel<<<dim3(131072), blk, 0, stream>>>(attn);
  attnv_kernel<<<dim3(4, 1, 128), blk, 0, stream>>>(attn, Vt, attnV);
  gemm1024<2><<<dim3(8, 64), blk, 0, stream>>>(attnV, Wo, bo, out);
}
